// Attention_edgeinfo_16681652977881
// MI455X (gfx1250) — compile-verified, matrix operand order checked by reading, not run on hardware
//
#include <hip/hip_runtime.h>
#include <hip/hip_bf16.h>

typedef __attribute__((ext_vector_type(16))) _Float16 v16h;
typedef __attribute__((ext_vector_type(8)))  float    v8f;
typedef __attribute__((ext_vector_type(4)))  unsigned int u32x4;
typedef __attribute__((ext_vector_type(8)))  int          i32x8;
typedef __attribute__((ext_vector_type(4)))  int          i32x4;

#define NB     4
#define SEQ    2048
#define DIMM   512
#define NHEAD  8
#define DHEAD  64
#define INNER  512
#define NQKV   1536
#define ATT_SCALE 0.125f

// ---------------------------------------------------------------------------
// TDM: DMA a 2D f16 tile (tile_rows x tile_cols) from a row-major tensor
// (tensor_rows x tile_cols, row stride = row_stride elems) into LDS.
// D# packing per CDNA5 ISA ch.8 (group0 128b, group1 256b; groups 2/3 zero
// for 2D tensors).  Tracked by TENSORcnt.
// ---------------------------------------------------------------------------
__device__ __forceinline__ void tdm_load_tile_f16(unsigned lds_byte_addr,
                                                  const void* gaddr,
                                                  unsigned tile_rows,
                                                  unsigned tile_cols,
                                                  unsigned tensor_rows,
                                                  unsigned row_stride) {
    unsigned long long ga = (unsigned long long)(uintptr_t)gaddr;
    u32x4 g0;
    g0[0] = 1u;                                           // count=1, user mode
    g0[1] = lds_byte_addr;                                // lds_addr
    g0[2] = (unsigned)(ga & 0xFFFFFFFFu);                 // global_addr[31:0]
    g0[3] = (unsigned)((ga >> 32) & 0x01FFFFFFu)          // global_addr[56:32]
          | (2u << 30);                                   // type = 2 ("image")
    i32x8 g1;
    g1[0] = (int)(1u << 16);                              // data_size=1 (2B), wg_mask=0
    g1[1] = (int)((tile_cols & 0xFFFFu) << 16);           // tensor_dim0[15:0]
    g1[2] = (int)(((tile_cols >> 16) & 0xFFFFu)           // tensor_dim0[31:16]
          |       ((tensor_rows & 0xFFFFu) << 16));       // tensor_dim1[15:0]
    g1[3] = (int)(((tensor_rows >> 16) & 0xFFFFu)         // tensor_dim1[31:16]
          |       ((tile_cols & 0xFFFFu) << 16));         // tile_dim0
    g1[4] = (int)(tile_rows & 0xFFFFu);                   // tile_dim1 (tile_dim2=0)
    g1[5] = (int)row_stride;                              // tensor_dim0_stride[31:0]
    g1[6] = 0;                                            // stride[47:32], dim1_stride lo
    g1[7] = 0;
    i32x4 z4 = {0, 0, 0, 0};
#if __has_include(<hip/amd_detail/amd_gfx1250_TDM.h>)
    i32x8 z8 = {0, 0, 0, 0, 0, 0, 0, 0};
    __builtin_amdgcn_tensor_load_to_lds(g0, g1, z4, z4, z8, 0);
#else
    __builtin_amdgcn_tensor_load_to_lds(g0, g1, z4, z4, 0);
#endif
}

// ---------------------------------------------------------------------------
// Kernel 1: QKV projection.  C[8192,1536] = x[8192,512] @ w_qkv[512,1536]
// Output scattered as f16 q/k/v tensors with layout [b, h, n, d].
// ---------------------------------------------------------------------------
__global__ void qkv_proj_kernel(const float* __restrict__ x,
                                const float* __restrict__ w,
                                _Float16* __restrict__ qo,
                                _Float16* __restrict__ ko,
                                _Float16* __restrict__ vo) {
    const int lane = threadIdx.x & 31;
    const int wave = threadIdx.x >> 5;
    const int lg   = lane & 15;
    const int hi8  = (lane & 16) ? 8 : 0;

    const int mt = blockIdx.x;                 // 0..511  (M = 8192)
    const int nt = blockIdx.y * 4 + wave;      // 0..95   (N = 1536)
    const int m0 = mt * 16;
    const int n0 = nt * 16;
    const int mrow = m0 + lg;
    const int ncol = n0 + lg;

    const float* arow = x + (size_t)mrow * DIMM;

    v8f acc = {};
    for (int kc = 0; kc < DIMM; kc += 32) {
        v16h a, b;
        const int kb = kc + hi8;
        #pragma unroll
        for (int e = 0; e < 8; ++e) {
            a[e]     = (_Float16)arow[kb + e];
            a[e + 8] = (_Float16)arow[kb + 16 + e];
        }
        #pragma unroll
        for (int e = 0; e < 16; ++e) {
            const int kk = kb + (e < 8 ? e : e + 8);
            b[e] = (_Float16)w[(size_t)kk * NQKV + ncol];
        }
        acc = __builtin_amdgcn_wmma_f32_16x16x32_f16(
            false, a, false, b, (short)0, acc, false, false);
    }

    const int sel = ncol >> 9;          // 0=q 1=k 2=v (tile never straddles)
    const int rem = ncol & 511;
    const int h   = rem >> 6;
    const int d   = rem & 63;
    _Float16* dst = (sel == 0) ? qo : ((sel == 1) ? ko : vo);
    #pragma unroll
    for (int r = 0; r < 8; ++r) {
        const int mm = m0 + r + hi8;
        const int bb = mm >> 11;
        const int nn = mm & (SEQ - 1);
        dst[(((size_t)bb * NHEAD + h) * SEQ + nn) * DHEAD + d] = (_Float16)acc[r];
    }
}

// ---------------------------------------------------------------------------
// Kernel 2: fused flash attention with additive bias (dist streamed once).
// 4 waves / block share one (b,h); K/V 32x64 tiles staged in LDS by the
// Tensor Data Mover, double-buffered; each wave owns a 16-row query tile.
// ---------------------------------------------------------------------------
__global__ void flash_attn_kernel(const float* __restrict__ dist,
                                  const _Float16* __restrict__ qg,
                                  const _Float16* __restrict__ kg,
                                  const _Float16* __restrict__ vg,
                                  _Float16* __restrict__ og) {
    __shared__ _Float16 kt[2][32][DHEAD];    // 8 KB: K tiles (double buffer)
    __shared__ _Float16 vt[2][32][DHEAD];    // 8 KB: V tiles (double buffer)
    __shared__ _Float16 lds_p[4][16][40];    // per-wave P tile (C->A relayout)

    const int lane = threadIdx.x & 31;
    const int wave = threadIdx.x >> 5;
    const int lg   = lane & 15;
    const int hi8  = (lane & 16) ? 8 : 0;

    const int bh = blockIdx.x >> 5;                   // 0..31  (b*8+h)
    const int b  = bh >> 3;
    const int h  = bh & 7;
    const int qt = ((blockIdx.x & 31) << 2) + wave;   // 0..127
    const int q0 = qt * 16;

    const _Float16* qbase = qg + ((size_t)bh * SEQ + q0) * DHEAD;
    const _Float16* kbase = kg + (size_t)bh * SEQ * DHEAD;
    const _Float16* vbase = vg + (size_t)bh * SEQ * DHEAD;
    const float*    dbase = dist + (size_t)bh * SEQ * SEQ;

    // Q fragments: A-layout, two K=32 chunks over d=64.
    v16h qf[2];
    #pragma unroll
    for (int c = 0; c < 2; ++c) {
        const _Float16* row = qbase + (size_t)lg * DHEAD;
        const int kb = c * 32 + hi8;
        #pragma unroll
        for (int e = 0; e < 8; ++e) {
            qf[c][e]     = row[kb + e];
            qf[c][e + 8] = row[kb + 16 + e];
        }
    }

    // Prologue: DMA first K/V tiles into buffer 0.
    if (wave == 0) {
        tdm_load_tile_f16((unsigned)(uintptr_t)&kt[0][0][0], kbase, 32, DHEAD, SEQ, DHEAD);
        tdm_load_tile_f16((unsigned)(uintptr_t)&vt[0][0][0], vbase, 32, DHEAD, SEQ, DHEAD);
        __builtin_amdgcn_s_wait_tensorcnt(0);
    }
    __syncthreads();

    const v8f zero = {};
    v8f oacc[4];
    #pragma unroll
    for (int c = 0; c < 4; ++c) oacc[c] = zero;
    float mrun[8], lrun[8];
    #pragma unroll
    for (int r = 0; r < 8; ++r) { mrun[r] = -1.0e30f; lrun[r] = 0.0f; }

    for (int j0 = 0; j0 < SEQ; j0 += 32) {
        const int buf = (j0 >> 5) & 1;

        // Issue next tile's DMA (overlaps with this iteration's compute).
        if (wave == 0 && (j0 + 32) < SEQ) {
            const int nb = buf ^ 1;
            tdm_load_tile_f16((unsigned)(uintptr_t)&kt[nb][0][0],
                              kbase + (size_t)(j0 + 32) * DHEAD, 32, DHEAD, SEQ, DHEAD);
            tdm_load_tile_f16((unsigned)(uintptr_t)&vt[nb][0][0],
                              vbase + (size_t)(j0 + 32) * DHEAD, 32, DHEAD, SEQ, DHEAD);
        }

        // ---- scores: two 16x16 tiles (keys j0..+15, j0+16..+31) ----
        v8f sc[2];
        sc[0] = zero; sc[1] = zero;
        #pragma unroll
        for (int s = 0; s < 2; ++s) {
            const _Float16* krow = &kt[buf][s * 16 + lg][0];
            #pragma unroll
            for (int c = 0; c < 2; ++c) {
                v16h bf;
                const int kb = c * 32 + hi8;
                #pragma unroll
                for (int e = 0; e < 8; ++e) {
                    bf[e]     = krow[kb + e];
                    bf[e + 8] = krow[kb + 16 + e];
                }
                sc[s] = __builtin_amdgcn_wmma_f32_16x16x32_f16(
                    false, qf[c], false, bf, (short)0, sc[s], false, false);
            }
        }

        // ---- scale + additive bias (the 536 MB stream) ----
        #pragma unroll
        for (int s = 0; s < 2; ++s) {
            const int col = j0 + s * 16 + lg;
            #pragma unroll
            for (int r = 0; r < 8; ++r) {
                const int row = q0 + r + hi8;
                sc[s][r] = sc[s][r] * ATT_SCALE + dbase[(size_t)row * SEQ + col];
            }
        }
        if (j0 + 32 < SEQ) {   // prefetch next bias tile -> global_prefetch_b8
            __builtin_prefetch(&dbase[(size_t)(q0 + hi8 + (lane & 7)) * SEQ + j0 + 32], 0, 1);
        }

        // ---- online softmax (row fixed per (r,hi8); N across 16 lanes) ----
        float tmax[8], corr[8], psum[8];
        #pragma unroll
        for (int r = 0; r < 8; ++r) tmax[r] = fmaxf(sc[0][r], sc[1][r]);
        #pragma unroll
        for (int m = 1; m <= 8; m <<= 1)
            #pragma unroll
            for (int r = 0; r < 8; ++r)
                tmax[r] = fmaxf(tmax[r], __shfl_xor(tmax[r], m, 32));

        #pragma unroll
        for (int r = 0; r < 8; ++r) {
            const float mnew = fmaxf(mrun[r], tmax[r]);
            corr[r] = __expf(mrun[r] - mnew);
            const float p0 = __expf(sc[0][r] - mnew);
            const float p1 = __expf(sc[1][r] - mnew);
            sc[0][r] = p0; sc[1][r] = p1;
            psum[r] = p0 + p1;
            mrun[r] = mnew;
        }
        #pragma unroll
        for (int m = 1; m <= 8; m <<= 1)
            #pragma unroll
            for (int r = 0; r < 8; ++r)
                psum[r] += __shfl_xor(psum[r], m, 32);
        #pragma unroll
        for (int r = 0; r < 8; ++r) lrun[r] = lrun[r] * corr[r] + psum[r];
        #pragma unroll
        for (int c = 0; c < 4; ++c)
            #pragma unroll
            for (int r = 0; r < 8; ++r) oacc[c][r] *= corr[r];

        // ---- P: C-layout -> A-layout via per-wave LDS tile ----
        #pragma unroll
        for (int s = 0; s < 2; ++s)
            #pragma unroll
            for (int r = 0; r < 8; ++r)
                lds_p[wave][r + hi8][s * 16 + lg] = (_Float16)sc[s][r];
        __syncthreads();

        v16h pf;
        {
            const _Float16* prow = &lds_p[wave][lg][0];
            #pragma unroll
            for (int e = 0; e < 8; ++e) {
                pf[e]     = prow[hi8 + e];
                pf[e + 8] = prow[hi8 + 16 + e];
            }
        }

        // ---- O += P @ V  (4 d-chunks of 16) ----
        #pragma unroll
        for (int c = 0; c < 4; ++c) {
            v16h vf;
            const int nd = c * 16 + lg;
            #pragma unroll
            for (int e = 0; e < 16; ++e) {
                const int kkey = hi8 + (e < 8 ? e : e + 8);
                vf[e] = vt[buf][kkey][nd];
            }
            oacc[c] = __builtin_amdgcn_wmma_f32_16x16x32_f16(
                false, pf, false, vf, (short)0, oacc[c], false, false);
        }

        // Drain next tile's DMA before the buffer swap, then block barrier.
        if (wave == 0) __builtin_amdgcn_s_wait_tensorcnt(0);
        __syncthreads();
    }

    // ---- epilogue: normalize and store O as f16 [b, n, h, d] ----
    #pragma unroll
    for (int c = 0; c < 4; ++c) {
        #pragma unroll
        for (int r = 0; r < 8; ++r) {
            const int row = q0 + r + hi8;
            og[(((size_t)b * SEQ + row) * NHEAD + h) * DHEAD + c * 16 + lg] =
                (_Float16)(oacc[c][r] / lrun[r]);
        }
    }
}

// ---------------------------------------------------------------------------
// Kernel 3: output projection. out[8192,512] = O_f16[8192,512] @ w_out + bias
// ---------------------------------------------------------------------------
__global__ void out_proj_kernel(const _Float16* __restrict__ og,
                                const float* __restrict__ w,
                                const float* __restrict__ bias,
                                float* __restrict__ out) {
    const int lane = threadIdx.x & 31;
    const int wave = threadIdx.x >> 5;
    const int lg   = lane & 15;
    const int hi8  = (lane & 16) ? 8 : 0;

    const int mt = blockIdx.x;               // 0..511
    const int nt = blockIdx.y * 4 + wave;    // 0..31
    const int m0 = mt * 16;
    const int n0 = nt * 16;
    const int mrow = m0 + lg;
    const int ncol = n0 + lg;

    const _Float16* arow = og + (size_t)mrow * INNER;

    v8f acc = {};
    for (int kc = 0; kc < INNER; kc += 32) {
        v16h a, b;
        const int kb = kc + hi8;
        #pragma unroll
        for (int e = 0; e < 8; ++e) {
            a[e]     = arow[kb + e];
            a[e + 8] = arow[kb + 16 + e];
        }
        #pragma unroll
        for (int e = 0; e < 16; ++e) {
            const int kk = kb + (e < 8 ? e : e + 8);
            b[e] = (_Float16)w[(size_t)kk * DIMM + ncol];
        }
        acc = __builtin_amdgcn_wmma_f32_16x16x32_f16(
            false, a, false, b, (short)0, acc, false, false);
    }

    const float bn = bias[ncol];
    #pragma unroll
    for (int r = 0; r < 8; ++r) {
        const int mm = m0 + r + hi8;
        out[(size_t)mm * DIMM + ncol] = acc[r] + bn;
    }
}

// ---------------------------------------------------------------------------
extern "C" void kernel_launch(void* const* d_in, const int* in_sizes, int n_in,
                              void* d_out, int out_size, void* d_ws, size_t ws_size,
                              hipStream_t stream) {
    (void)in_sizes; (void)n_in; (void)out_size; (void)ws_size;

    const float* x     = (const float*)d_in[0];
    const float* dist  = (const float*)d_in[1];
    const float* w_qkv = (const float*)d_in[2];
    const float* w_out = (const float*)d_in[3];
    const float* b_out = (const float*)d_in[4];
    float* out = (float*)d_out;

    const size_t per = (size_t)NB * NHEAD * SEQ * DHEAD;   // 4,194,304 elems
    _Float16* qws = (_Float16*)d_ws;
    _Float16* kws = qws + per;
    _Float16* vws = kws + per;
    _Float16* ows = vws + per;   // total 32 MiB of d_ws

    dim3 blk(128, 1, 1);   // 4 wave32s

    qkv_proj_kernel<<<dim3(512, 24, 1), blk, 0, stream>>>(x, w_qkv, qws, kws, vws);
    flash_attn_kernel<<<dim3(1024, 1, 1), blk, 0, stream>>>(dist, qws, kws, vws, ows);
    out_proj_kernel<<<dim3(512, 8, 1), blk, 0, stream>>>(ows, w_out, b_out, out);
}
